// GNNStructEncoder_80599356276736
// MI455X (gfx1250) — compile-verified, matrix operand order checked by reading, not run on hardware
//
#include <hip/hip_runtime.h>
#include <hip/hip_bf16.h>
#include <math.h>

// ---------------- problem constants ----------------
#define NNODE 20000
#define NEDGE 640000
#define DHID  256
#define SSAMP 10
#define NBF   ((size_t)NNODE * DHID)          // floats per activation matrix
#define NBBYTES (NBF * sizeof(float))
#define WELEM (DHID * DHID)                    // 65536 elems per weight matrix
#define WBYTES_BF16 (WELEM * 2)                // 131072 B packed bf16 weights
#define ROWS_PER_ITER 32                       // 2 row-subtiles x 16 rows
#define TILES_PER_BLOCK 5                      // 625 row-pairs / 125 blocks

typedef __attribute__((ext_vector_type(16))) __bf16 v16bf;
typedef __attribute__((ext_vector_type(8)))  float  v8f;

__device__ __forceinline__ float wave_sum(float v) {
    #pragma unroll
    for (int off = 16; off > 0; off >>= 1) v += __shfl_xor(v, off, 32);
    return v;
}

// ---------------- weight pack: f32 row-major [K=256][N=256] -> WMMA-B fragment order ----
// packed index = ((nt*8 + kt)*512 + lane*16 + e), nt in [0,16), kt in [0,8)
// lane: n = lane&15, half = lane>>4 ; element e maps K = kt*32 + (e<8 ? half*8+e : 16+half*8+e-8)
__global__ void pack_w_kernel(const float* __restrict__ W, __bf16* __restrict__ out) {
    int tid  = blockIdx.x * 256 + threadIdx.x;          // 65536 total
    int frag = tid >> 9;
    int within = tid & 511;
    int lane = within >> 4;
    int e    = within & 15;
    int nt   = frag >> 3;
    int kt   = frag & 7;
    int half = lane >> 4;
    int n    = lane & 15;
    int kk   = kt * 32 + (e < 8 ? half * 8 + e : 16 + half * 8 + (e - 8));
    out[tid] = (__bf16)W[kk * DHID + nt * 16 + n];
}

// ---------------- persistent WMMA GEMM: Y = act(X @ W + b) ----------------
// Block = 256 threads (8 waves). Stage whole packed weight matrix (128 KB bf16)
// into LDS once via gfx1250 async global->LDS copy, then process
// TILES_PER_BLOCK iterations of 32 rows x 256 cols. Wave w: row-subtile w>>2,
// column group (w&3)*64 (4 WMMA accumulators per wave).
__global__ __launch_bounds__(256)
void gemm_bias_kernel(const float* __restrict__ X, const __bf16* __restrict__ Wp,
                      const float* __restrict__ bias, float* __restrict__ Y, int relu) {
    extern __shared__ char smem[];                      // 131072 B dynamic LDS
    int tid  = threadIdx.x;
    int lane = tid & 31;
    int wave = tid >> 5;
    int half = lane >> 4;
    int mcol = lane & 15;
    int rowgrp = wave >> 2;            // 0..1
    int ntbase = (wave & 3) * 4;       // 4 col tiles of 16 -> 64 cols

    // ---- stage packed weights into LDS with async global->LDS copies ----
    {
        unsigned long long gbase = (unsigned long long)(const void*)Wp;
        #pragma unroll 4
        for (int it = 0; it < (int)(WBYTES_BF16 / (256 * 16)); ++it) {   // 32 iters
            unsigned lds_off = (unsigned)(tid * 16 + it * 4096);
            unsigned long long gaddr = gbase + (unsigned long long)lds_off;
            asm volatile("global_load_async_to_lds_b128 %0, %1, off"
                         :: "v"(lds_off), "v"(gaddr) : "memory");
        }
        asm volatile("s_wait_asynccnt 0x0" ::: "memory");
        __syncthreads();
    }

    for (int t = 0; t < TILES_PER_BLOCK; ++t) {
        int tile = blockIdx.x * TILES_PER_BLOCK + t;
        size_t row = (size_t)tile * ROWS_PER_ITER + rowgrp * 16 + mcol;
        const float* xrow = X + row * DHID;

        v8f acc0 = {}, acc1 = {}, acc2 = {}, acc3 = {};

        #pragma unroll
        for (int kt = 0; kt < 8; ++kt) {
            int kb = kt * 32 + half * 8;
            float4 a0 = *(const float4*)(xrow + kb);
            float4 a1 = *(const float4*)(xrow + kb + 4);
            float4 a2 = *(const float4*)(xrow + kb + 16);
            float4 a3 = *(const float4*)(xrow + kb + 20);
            v16bf a;
            a[0]=(__bf16)a0.x;  a[1]=(__bf16)a0.y;  a[2]=(__bf16)a0.z;  a[3]=(__bf16)a0.w;
            a[4]=(__bf16)a1.x;  a[5]=(__bf16)a1.y;  a[6]=(__bf16)a1.z;  a[7]=(__bf16)a1.w;
            a[8]=(__bf16)a2.x;  a[9]=(__bf16)a2.y;  a[10]=(__bf16)a2.z; a[11]=(__bf16)a2.w;
            a[12]=(__bf16)a3.x; a[13]=(__bf16)a3.y; a[14]=(__bf16)a3.z; a[15]=(__bf16)a3.w;

            // B fragments from LDS: frag (nt*8+kt) is 1024 B, lane slice is 32 B
            const char* fb = smem + (size_t)((ntbase * 8 + kt) * 1024) + (size_t)lane * 32;
            v16bf b0 = *(const v16bf*)(fb);
            v16bf b1 = *(const v16bf*)(fb + 8 * 1024);
            v16bf b2 = *(const v16bf*)(fb + 16 * 1024);
            v16bf b3 = *(const v16bf*)(fb + 24 * 1024);

            acc0 = __builtin_amdgcn_wmma_f32_16x16x32_bf16(false, a, false, b0, (short)0, acc0, false, false);
            acc1 = __builtin_amdgcn_wmma_f32_16x16x32_bf16(false, a, false, b1, (short)0, acc1, false, false);
            acc2 = __builtin_amdgcn_wmma_f32_16x16x32_bf16(false, a, false, b2, (short)0, acc2, false, false);
            acc3 = __builtin_amdgcn_wmma_f32_16x16x32_bf16(false, a, false, b3, (short)0, acc3, false, false);
        }

        size_t rb = (size_t)tile * ROWS_PER_ITER + rowgrp * 16 + half * 8;
        #pragma unroll
        for (int j = 0; j < 4; ++j) {
            v8f acc = (j == 0) ? acc0 : (j == 1) ? acc1 : (j == 2) ? acc2 : acc3;
            int col = (ntbase + j) * 16 + mcol;
            float bv = bias[col];
            #pragma unroll
            for (int r = 0; r < 8; ++r) {
                float v = acc[r] + bv;
                if (relu) v = fmaxf(v, 0.0f);
                Y[(rb + r) * DHID + col] = v;
            }
        }
    }
}

// ---------------- agg = copy then scatter-add over edges ----------------
__global__ void copy4_kernel(const float4* __restrict__ s, float4* __restrict__ d) {
    size_t i = (size_t)blockIdx.x * 256 + threadIdx.x;
    d[i] = s[i];
}

__global__ void scatter_add_kernel(const float* __restrict__ feat, const int* __restrict__ src,
                                   const int* __restrict__ dst, float* __restrict__ agg) {
    int e = blockIdx.x * 4 + (threadIdx.x >> 6);       // 64 threads per edge
    if (e >= NEDGE) return;
    int d = (threadIdx.x & 63) * 4;
    int s = src[e], t = dst[e];
    float4 v = *(const float4*)(feat + (size_t)s * DHID + d);
    float* a = agg + (size_t)t * DHID + d;
    atomicAdd(a + 0, v.x); atomicAdd(a + 1, v.y);
    atomicAdd(a + 2, v.z); atomicAdd(a + 3, v.w);
}

// ---------------- row-wise l2 normalize (wave per row) ----------------
__global__ void l2norm_kernel(const float* __restrict__ in, float* __restrict__ out) {
    int lane = threadIdx.x & 31, wave = threadIdx.x >> 5;
    size_t n = (size_t)blockIdx.x * 8 + wave;
    const float4* r = (const float4*)(in + n * DHID) + lane * 2;
    float4 x0 = r[0], x1 = r[1];
    float s = x0.x*x0.x + x0.y*x0.y + x0.z*x0.z + x0.w*x0.w
            + x1.x*x1.x + x1.y*x1.y + x1.z*x1.z + x1.w*x1.w;
    s = wave_sum(s);
    float inv = 1.0f / fmaxf(sqrtf(s), 1e-12f);
    float4* w = (float4*)(out + n * DHID) + lane * 2;
    float4 y0, y1;
    y0.x=x0.x*inv; y0.y=x0.y*inv; y0.z=x0.z*inv; y0.w=x0.w*inv;
    y1.x=x1.x*inv; y1.y=x1.y*inv; y1.z=x1.z*inv; y1.w=x1.w*inv;
    w[0] = y0; w[1] = y1;
}

// ---------------- contrastive loss per node (wave per node) ----------------
__global__ void loss_kernel(const float* __restrict__ proj, const float* __restrict__ tgt,
                            const float* __restrict__ onorm, const int* __restrict__ pos_idx,
                            const int* __restrict__ neg_idx, float* __restrict__ loss_n) {
    int lane = threadIdx.x & 31, wave = threadIdx.x >> 5;
    int n = blockIdx.x * 8 + wave;
    const float4* pr = (const float4*)(proj + (size_t)n * DHID) + lane * 2;
    float4 p0 = pr[0], p1 = pr[1];
    const float4* orow = (const float4*)(onorm + (size_t)n * DHID) + lane * 2;
    float4 o0 = orow[0], o1 = orow[1];
    float pos = 0.0f, neg = 0.0f;
    for (int s = 0; s < SSAMP; ++s) {
        int pi = pos_idx[n * SSAMP + s];
        const float4* tr = (const float4*)(tgt + (size_t)pi * DHID) + lane * 2;
        float4 t0 = tr[0], t1 = tr[1];
        float d = p0.x*t0.x + p0.y*t0.y + p0.z*t0.z + p0.w*t0.w
                + p1.x*t1.x + p1.y*t1.y + p1.z*t1.z + p1.w*t1.w;
        d = wave_sum(d);
        pos += expf(d * 2.0f);            // /TAU, TAU=0.5
        int ni = neg_idx[n * SSAMP + s];
        const float4* nr = (const float4*)(onorm + (size_t)ni * DHID) + lane * 2;
        float4 m0 = nr[0], m1 = nr[1];
        float dn = o0.x*m0.x + o0.y*m0.y + o0.z*m0.z + o0.w*m0.w
                 + o1.x*m1.x + o1.y*m1.y + o1.z*m1.z + o1.w*m1.w;
        dn = wave_sum(dn);
        neg += expf(dn * 2.0f);
    }
    if (lane == 0) loss_n[n] = -logf(pos / (pos + neg));
}

__global__ void reduce_loss_kernel(const float* __restrict__ loss_n, float* __restrict__ out) {
    __shared__ float sm[256];
    float s = 0.0f;
    for (int i = threadIdx.x; i < NNODE; i += 256) s += loss_n[i];
    sm[threadIdx.x] = s;
    __syncthreads();
    for (int st = 128; st > 0; st >>= 1) {
        if ((int)threadIdx.x < st) sm[threadIdx.x] += sm[threadIdx.x + st];
        __syncthreads();
    }
    if (threadIdx.x == 0) out[0] = sm[0] / (float)NNODE;
}

__global__ void copy_out_kernel(const float* __restrict__ src, float* __restrict__ dst) {
    size_t i = (size_t)blockIdx.x * 256 + threadIdx.x;   // dst may be 4B-aligned only
    dst[i] = src[i];
}

// ---------------- host orchestration ----------------
extern "C" void kernel_launch(void* const* d_in, const int* in_sizes, int n_in,
                              void* d_out, int out_size, void* d_ws, size_t ws_size,
                              hipStream_t stream) {
    const float* h       = (const float*)d_in[0];
    const int*   src     = (const int*)d_in[1];
    const int*   dst     = (const int*)d_in[2];
    const int*   pos_idx = (const int*)d_in[3];
    const int*   neg_idx = (const int*)d_in[4];
    // weights/biases in dict order
    const void* w11 = d_in[5];  const float* b11 = (const float*)d_in[6];
    const void* w12 = d_in[7];  const float* b12 = (const float*)d_in[8];
    const void* w21 = d_in[9];  const float* b21 = (const float*)d_in[10];
    const void* w22 = d_in[11]; const float* b22 = (const float*)d_in[12];
    const void* tw11 = d_in[13]; const float* tb11 = (const float*)d_in[14];
    const void* tw12 = d_in[15]; const float* tb12 = (const float*)d_in[16];
    const void* tw21 = d_in[17]; const float* tb21 = (const float*)d_in[18];
    const void* tw22 = d_in[19]; const float* tb22 = (const float*)d_in[20];
    const void* pw1 = d_in[21]; const float* pb1 = (const float*)d_in[22];
    const void* pw2 = d_in[23]; const float* pb2 = (const float*)d_in[24];
    const void* pw3 = d_in[25]; const float* pb3 = (const float*)d_in[26];

    char* ws = (char*)d_ws;
    float* B0 = (float*)(ws + 0 * NBBYTES);
    float* B1 = (float*)(ws + 1 * NBBYTES);
    float* B2 = (float*)(ws + 2 * NBBYTES);
    float* B3 = (float*)(ws + 3 * NBBYTES);   // output_emb
    float* B4 = (float*)(ws + 4 * NBBYTES);   // out_norm
    __bf16* PW = (__bf16*)(ws + 5 * NBBYTES);
    float* LOSSN = (float*)(ws + 5 * NBBYTES + (size_t)11 * WELEM * sizeof(__bf16));

    // pack all 11 weight matrices to bf16 WMMA-fragment layout
    const void* wsrc[11] = { w11, w12, w21, w22, tw11, tw12, tw21, tw22, pw1, pw2, pw3 };
    for (int i = 0; i < 11; ++i)
        pack_w_kernel<<<256, 256, 0, stream>>>((const float*)wsrc[i], PW + (size_t)i * WELEM);

    const int gemm_grid = NNODE / ROWS_PER_ITER / TILES_PER_BLOCK;  // 125 persistent blocks
    const int copy_grid = (int)(NBF / 4 / 256);                     // 5000
    const int scat_grid = NEDGE / 4;                                // 160000

    auto GEMM = [&](const float* X, int wslot, const float* bias, float* Y, int relu) {
        gemm_bias_kernel<<<gemm_grid, 256, WBYTES_BF16, stream>>>(
            X, PW + (size_t)wslot * WELEM, bias, Y, relu);
    };
    auto AGG = [&](const float* feat, float* agg) {
        copy4_kernel<<<copy_grid, 256, 0, stream>>>((const float4*)feat, (float4*)agg);
        scatter_add_kernel<<<scat_grid, 256, 0, stream>>>(feat, src, dst, agg);
    };

    // agg0 = h + scatter(h)   (shared by online + target layer 1)
    AGG(h, B0);
    // online GIN layer 1: l1 = MLP(agg0)
    GEMM(B0, 0, b11, B1, 1);
    GEMM(B1, 1, b12, B2, 0);                 // B2 = l1
    // agg1 = l1 + scatter(l1)
    AGG(B2, B1);
    // online GIN layer 2: output_emb
    GEMM(B1, 2, b21, B2, 1);
    GEMM(B2, 3, b22, B3, 0);                 // B3 = output_emb
    // target GIN layer 1 from agg0
    GEMM(B0, 4, tb11, B1, 1);
    GEMM(B1, 5, tb12, B2, 0);                // B2 = t1
    // aggt = t1 + scatter(t1)
    AGG(B2, B0);
    // target GIN layer 2 -> t2 -> target_emb (l2norm in place)
    GEMM(B0, 6, tb21, B1, 1);
    GEMM(B1, 7, tb22, B2, 0);
    l2norm_kernel<<<NNODE / 8, 256, 0, stream>>>(B2, B2);   // B2 = target_emb
    // projector: 3 layers -> projected_rep (l2norm in place)
    GEMM(B3, 8, pb1, B0, 1);
    GEMM(B0, 9, pb2, B1, 1);
    GEMM(B1, 10, pb3, B0, 0);
    l2norm_kernel<<<NNODE / 8, 256, 0, stream>>>(B0, B0);   // B0 = projected_rep
    // out_norm
    l2norm_kernel<<<NNODE / 8, 256, 0, stream>>>(B3, B4);   // B4 = out_norm
    // contrastive loss
    loss_kernel<<<NNODE / 8, 256, 0, stream>>>(B0, B2, B4, pos_idx, neg_idx, LOSSN);
    reduce_loss_kernel<<<1, 256, 0, stream>>>(LOSSN, (float*)d_out);
    // output_emb -> d_out[1:]
    copy_out_kernel<<<NNODE * DHID / 256, 256, 0, stream>>>(B3, (float*)d_out + 1);
}
